// MultiHeadAttention_35021163331760
// MI455X (gfx1250) — matrix operand order repaired and checked by reading, not compiled, not run on hardware
//
#include <hip/hip_runtime.h>
#include <math.h>
#include <stdint.h>

#define H_  8
#define D_  512
#define DK_ 64
#define DV_ 64
#define B_  2
#define L_  2048
#define LN_EPS 1e-5f

typedef __attribute__((ext_vector_type(16))) __bf16        v16bf;
typedef __attribute__((ext_vector_type(8)))  float         v8f;
typedef __attribute__((ext_vector_type(4)))  unsigned int  u32x4;
typedef __attribute__((ext_vector_type(4)))  int           i32x4v;
typedef __attribute__((ext_vector_type(4)))  float         f32x4;

union FragU {
    v16bf          v;
    u32x4          q[2];
    unsigned int   u[8];
    unsigned short s[16];
};

// round-to-nearest-even bf16, kept in the high 16 bits of a dword
__device__ __forceinline__ unsigned int bfround(float f) {
    unsigned int u = __float_as_uint(f);
    return u + 0x7FFFu + ((u >> 16) & 1u);
}
__device__ __forceinline__ unsigned short f2bf(float f) {
    return (unsigned short)(bfround(f) >> 16);
}
// pack two floats -> {bf16(hi), bf16(lo)} with one v_perm_b32
__device__ __forceinline__ unsigned int pk2bf(float lo, float hi) {
    return __builtin_amdgcn_perm(bfround(hi), bfround(lo), 0x07060302u);
}

#define WMMA_BF16(A, Bm, C) \
    __builtin_amdgcn_wmma_f32_16x16x32_bf16(false, (A), false, (Bm), (short)0, (C), false, false)

// ---- CDNA5 async global->LDS copy (16B per lane), with sync fallback ------
#if __has_builtin(__builtin_amdgcn_global_load_async_to_lds_b128) && \
    __has_builtin(__builtin_amdgcn_s_wait_asynccnt)
#define USE_ASYNC 1
#define GAS1(p) ((__attribute__((address_space(1))) i32x4v*)(unsigned long long)(uintptr_t)(p))
#define LAS3(p) ((__attribute__((address_space(3))) i32x4v*)(unsigned int)(uintptr_t)(p))
#define CP16(ldst, gsrc) \
    __builtin_amdgcn_global_load_async_to_lds_b128(GAS1(gsrc), LAS3(ldst), 0, 0)
#define WAIT_ASYNC() __builtin_amdgcn_s_wait_asynccnt(0)
#else
#define USE_ASYNC 0
#define CP16(ldst, gsrc) (*(u32x4*)(ldst) = *(const u32x4*)(gsrc))
#define WAIT_ASYNC() ((void)0)
#endif

// ---------------------------------------------------------------------------
// One-pass f32 -> bf16 conversion (8 elements per thread).
// ---------------------------------------------------------------------------
__global__ __launch_bounds__(256) void cvt_bf16_kernel(const float* __restrict__ src,
                                                       unsigned short* __restrict__ dst,
                                                       int n8) {
    const int i = blockIdx.x * 256 + threadIdx.x;
    if (i >= n8) return;
    const f32x4* s = (const f32x4*)src + 2 * (size_t)i;
    f32x4 x0 = s[0], x1 = s[1];
    u32x4 o;
    o[0] = pk2bf(x0[0], x0[1]);
    o[1] = pk2bf(x0[2], x0[3]);
    o[2] = pk2bf(x1[0], x1[1]);
    o[3] = pk2bf(x1[2], x1[3]);
    *((u32x4*)dst + (size_t)i) = o;
}

// ---------------------------------------------------------------------------
// Projection: Y = Xb (4096x512 bf16) @ Wb^T (512x512 bf16), bf16 out.
// One wave = 16x64 output tile (4 accumulators, shared A fragment).
// mode 0: Q (scaled 1/8) -> [hb][L][64];  1: K -> [hb][L][64];  2: V^T -> [hb][64][L]
// Pure bf16 inner loop: 2 + 8 x b128 loads -> 4 WMMAs per k-step.
// ---------------------------------------------------------------------------
__global__ __launch_bounds__(128) void proj_kernel(const unsigned short* __restrict__ Xb,
                                                   const unsigned short* __restrict__ Wb,
                                                   unsigned short* __restrict__ OUT,
                                                   int mode) {
    const int lane = threadIdx.x & 31;
    const int wid  = blockIdx.x * 4 + (threadIdx.x >> 5);
    const int tile_m = wid >> 3;       // 256 row tiles
    const int ngrp   = wid & 7;        // 8 groups of 64 output cols == head id
    const int m0 = tile_m * 16, n0 = ngrp * 64;
    const int g  = lane >> 4;
    const int lm = lane & 15;

    const unsigned short* arow = Xb + (size_t)(m0 + lm) * D_ + g * 16;
    const unsigned short* brow[4];
#pragma unroll
    for (int t = 0; t < 4; ++t)
        brow[t] = Wb + (size_t)(n0 + t * 16 + lm) * D_ + g * 16;

    v8f acc[4] = {};
    for (int k0 = 0; k0 < D_; k0 += 32) {
        FragU a;
        a.q[0] = *(const u32x4*)(arow + k0);
        a.q[1] = *(const u32x4*)(arow + k0 + 8);
#pragma unroll
        for (int t = 0; t < 4; ++t) {
            FragU bm;
            bm.q[0] = *(const u32x4*)(brow[t] + k0);
            bm.q[1] = *(const u32x4*)(brow[t] + k0 + 8);
            acc[t] = WMMA_BF16(a.v, bm.v, acc[t]);
        }
    }

    // Scatter. h == ngrp for the whole wave. Hoisted mode branch.
    const int h = ngrp;
    const int b = m0 >> 11;
    const int lrow0 = (m0 & (L_ - 1)) + g * 8;
    const float scl = (mode == 0) ? 0.125f : 1.0f;   // fold 1/sqrt(DK) into Q
    size_t base; size_t rstride;
    if (mode < 2) {  // [hb][L][64]
        base = ((size_t)(h * B_ + b) * L_ + lrow0) * 64 + lm;
        rstride = 64;
    } else {         // [hb][64][L]
        base = ((size_t)(h * B_ + b) * 64 + lm) * L_ + lrow0;
        rstride = 1;
    }
    const size_t tstride = (mode < 2) ? (size_t)16 : (size_t)16 * L_;
#pragma unroll
    for (int t = 0; t < 4; ++t)
#pragma unroll
        for (int r = 0; r < 8; ++r)
            OUT[base + t * tstride + r * rstride] = f2bf(acc[t][r] * scl);
}

// ---------------------------------------------------------------------------
// Flash attention. One block = 4 waves on the SAME (hb), 4 consecutive q-tiles.
// K (32x64) and V^T (64x32) tiles staged once per block into double-buffered
// LDS via async global->LDS copies (ASYNCcnt pipeline), shared by all waves.
// ---------------------------------------------------------------------------
__global__ __launch_bounds__(128) void attn_kernel(const unsigned short* __restrict__ Qh,
                                                   const unsigned short* __restrict__ Kh,
                                                   const unsigned short* __restrict__ Vt,
                                                   const unsigned char* __restrict__ mask,
                                                   unsigned short* __restrict__ ctx) {
    __shared__ __align__(16) unsigned short sK[2][32][64];    // [buf][key][dk]
    __shared__ __align__(16) unsigned short sV[2][64][32];    // [buf][dv][key]
    __shared__ __align__(16) unsigned short lds_p[4][16][32]; // P transpose

    const int tid  = threadIdx.x;
    const int lane = tid & 31;
    const int w    = tid >> 5;
    const int hb   = blockIdx.x >> 5;            // 16 head-batches
    const int qt   = (blockIdx.x & 31) * 4 + w;  // 128 q tiles per hb
    const int h    = hb >> 1, b = hb & 1;
    const int g    = lane >> 4;
    const int lm   = lane & 15;

    const unsigned short* Kbase = Kh + (size_t)hb * L_ * DK_;
    const unsigned short* Vbase = Vt + (size_t)hb * DV_ * L_;

    // Resident Q fragments (dk 0..31 / 32..63), pre-scaled by 1/8.
    FragU aq[2];
    {
        const unsigned short* qp = Qh + ((size_t)hb * L_ + qt * 16 + lm) * DK_;
#pragma unroll
        for (int j = 0; j < 2; ++j) {
            const unsigned short* p = qp + j * 32 + g * 16;
            aq[j].q[0] = *(const u32x4*)(p);
            aq[j].q[1] = *(const u32x4*)(p + 8);
        }
    }

    v8f o[4] = {};
    float m_row[8], l_row[8];
#pragma unroll
    for (int r = 0; r < 8; ++r) { m_row[r] = -3.0e38f; l_row[r] = 0.0f; }

    const size_t maskbase = (size_t)hb * L_ * L_;

    auto stage = [&](int buf, int kt) {
#pragma unroll
        for (int c = 0; c < 2; ++c) {
            const int ck = tid + c * 128;
            const int krow = ck >> 3, koff = (ck & 7) * 8;
            CP16(&sK[buf][krow][koff], Kbase + ((size_t)(kt + krow)) * DK_ + koff);
            const int vrow = ck >> 2, voff = (ck & 3) * 8;
            CP16(&sV[buf][vrow][voff], Vbase + (size_t)vrow * L_ + kt + voff);
        }
    };

    stage(0, 0);

    for (int it = 0; it < L_ / 32; ++it) {
        const int kt  = it * 32;
        const int cur = it & 1;
        WAIT_ASYNC();
        __syncthreads();
        if (it + 1 < L_ / 32) stage(1 - cur, kt + 32);
        __builtin_prefetch(mask + maskbase + (size_t)(qt * 16) * L_ + kt + 32, 0, 0);

        // ---- scores S = (Q/8) @ K^T : two 16-col tiles ----
        v8f s[2];
#pragma unroll
        for (int st = 0; st < 2; ++st) {
            FragU bk0, bk1;
            const unsigned short* kp = &sK[cur][st * 16 + lm][g * 16];
            __builtin_memcpy(&bk0.q[0], kp, 16);
            __builtin_memcpy(&bk0.q[1], kp + 8, 16);
            __builtin_memcpy(&bk1.q[0], kp + 32, 16);
            __builtin_memcpy(&bk1.q[1], kp + 40, 16);
            v8f c = {};
            c = WMMA_BF16(aq[0].v, bk0.v, c);
            c = WMMA_BF16(aq[1].v, bk1.v, c);
            s[st] = c;
        }

        // ---- mask + online softmax ----
        const int q0 = qt * 16 + g * 8;
        float mnew[8];
#pragma unroll
        for (int r = 0; r < 8; ++r) {
            const unsigned char* mp = mask + maskbase + (size_t)(q0 + r) * L_ + kt;
            float s0 = mp[lm]      ? -1e10f : s[0][r];
            float s1 = mp[16 + lm] ? -1e10f : s[1][r];
            s[0][r] = s0; s[1][r] = s1;
            float t = fmaxf(s0, s1);
            t = fmaxf(t, __shfl_xor(t, 1, 32));
            t = fmaxf(t, __shfl_xor(t, 2, 32));
            t = fmaxf(t, __shfl_xor(t, 4, 32));
            t = fmaxf(t, __shfl_xor(t, 8, 32));
            mnew[r] = fmaxf(m_row[r], t);
        }
#pragma unroll
        for (int r = 0; r < 8; ++r) {
            const float alpha = __expf(m_row[r] - mnew[r]);
            const float p0 = __expf(s[0][r] - mnew[r]);
            const float p1 = __expf(s[1][r] - mnew[r]);
            float rs = p0 + p1;
            rs += __shfl_xor(rs, 1, 32);
            rs += __shfl_xor(rs, 2, 32);
            rs += __shfl_xor(rs, 4, 32);
            rs += __shfl_xor(rs, 8, 32);
            l_row[r] = l_row[r] * alpha + rs;
            m_row[r] = mnew[r];
#pragma unroll
            for (int t4 = 0; t4 < 4; ++t4) o[t4][r] *= alpha;
            lds_p[w][g * 8 + r][lm]      = f2bf(p0);
            lds_p[w][g * 8 + r][16 + lm] = f2bf(p1);
        }

        // ---- P (16x32) as A fragment (per-wave region) ----
        FragU ap;
        {
            const unsigned short* pp = &lds_p[w][lm][0];
            __builtin_memcpy(&ap.q[0], pp + g * 16, 16);
            __builtin_memcpy(&ap.q[1], pp + g * 16 + 8, 16);
        }

        // ---- O += P @ V : 4 col tiles of 16 ----
#pragma unroll
        for (int t4 = 0; t4 < 4; ++t4) {
            FragU bv;
            const unsigned short* vp = &sV[cur][t4 * 16 + lm][g * 16];
            __builtin_memcpy(&bv.q[0], vp, 16);
            __builtin_memcpy(&bv.q[1], vp + 8, 16);
            o[t4] = WMMA_BF16(ap.v, bv.v, o[t4]);
        }
    }

    // ---- normalize + write ctx (b,l, h*64+dv) bf16 ----
#pragma unroll
    for (int r = 0; r < 8; ++r) {
        const int qrow = qt * 16 + g * 8 + r;
        const float inv = 1.0f / l_row[r];
#pragma unroll
        for (int t4 = 0; t4 < 4; ++t4)
            ctx[((size_t)(b * L_ + qrow)) * (H_ * DV_) + h * DV_ + t4 * 16 + lm] =
                f2bf(o[t4][r] * inv);
    }
}

// ---------------------------------------------------------------------------
// FC + bias + residual + LayerNorm. One block = 16 full rows. Pure bf16 GEMM.
// ---------------------------------------------------------------------------
__global__ __launch_bounds__(256) void fc_ln_kernel(const unsigned short* __restrict__ ctx,
                                                    const unsigned short* __restrict__ Wfcb,
                                                    const float* __restrict__ bfc,
                                                    const float* __restrict__ resid,
                                                    const float* __restrict__ gamma,
                                                    const float* __restrict__ beta,
                                                    float* __restrict__ out) {
    __shared__ float s_sum[16], s_sq[16];
    const int lane = threadIdx.x & 31;
    const int w    = threadIdx.x >> 5;      // cols [w*64, w*64+64)
    const int m0   = blockIdx.x * 16;
    const int g    = lane >> 4;
    const int lm   = lane & 15;

    if (threadIdx.x < 16) { s_sum[threadIdx.x] = 0.0f; s_sq[threadIdx.x] = 0.0f; }
    __syncthreads();

    const unsigned short* arow = ctx + (size_t)(m0 + lm) * D_ + g * 16;
    const unsigned short* brow[4];
#pragma unroll
    for (int t4 = 0; t4 < 4; ++t4)
        brow[t4] = Wfcb + (size_t)(w * 64 + t4 * 16 + lm) * D_ + g * 16;

    v8f acc[4] = {};
    for (int k0 = 0; k0 < D_; k0 += 32) {
        FragU a;
        a.q[0] = *(const u32x4*)(arow + k0);
        a.q[1] = *(const u32x4*)(arow + k0 + 8);
#pragma unroll
        for (int t4 = 0; t4 < 4; ++t4) {
            FragU bm;
            bm.q[0] = *(const u32x4*)(brow[t4] + k0);
            bm.q[1] = *(const u32x4*)(brow[t4] + k0 + 8);
            acc[t4] = WMMA_BF16(a.v, bm.v, acc[t4]);
        }
    }

    float x[4][8];
#pragma unroll
    for (int t4 = 0; t4 < 4; ++t4) {
        const int n = w * 64 + t4 * 16 + lm;
        const float bias = bfc[n];
#pragma unroll
        for (int r = 0; r < 8; ++r) {
            const int m = m0 + g * 8 + r;
            x[t4][r] = acc[t4][r] + bias + resid[(size_t)m * D_ + n];
        }
    }

#pragma unroll
    for (int r = 0; r < 8; ++r) {
        float s1 = x[0][r] + x[1][r] + x[2][r] + x[3][r];
        float s2 = x[0][r] * x[0][r] + x[1][r] * x[1][r] +
                   x[2][r] * x[2][r] + x[3][r] * x[3][r];
        s1 += __shfl_xor(s1, 1, 32); s1 += __shfl_xor(s1, 2, 32);
        s1 += __shfl_xor(s1, 4, 32); s1 += __shfl_xor(s1, 8, 32);
        s2 += __shfl_xor(s2, 1, 32); s2 += __shfl_xor(s2, 2, 32);
        s2 += __shfl_xor(s2, 4, 32); s2 += __shfl_xor(s2, 8, 32);
        if (lm == 0) {
            atomicAdd(&s_sum[g * 8 + r], s1);
            atomicAdd(&s_sq[g * 8 + r], s2);
        }
    }
    __syncthreads();

#pragma unroll
    for (int r = 0; r < 8; ++r) {
        const int row   = g * 8 + r;
        const float mean = s_sum[row] * (1.0f / D_);
        const float var  = s_sq[row] * (1.0f / D_) - mean * mean;
        const float rinv = rsqrtf(var + LN_EPS);
#pragma unroll
        for (int t4 = 0; t4 < 4; ++t4) {
            const int n = w * 64 + t4 * 16 + lm;
            out[(size_t)(m0 + row) * D_ + n] = gamma[n] * (x[t4][r] - mean) * rinv + beta[n];
        }
    }
}

// ---------------------------------------------------------------------------
extern "C" void kernel_launch(void* const* d_in, const int* in_sizes, int n_in,
                              void* d_out, int out_size, void* d_ws, size_t ws_size,
                              hipStream_t stream) {
    (void)in_sizes; (void)n_in; (void)out_size; (void)ws_size;
    const float*         q     = (const float*)d_in[0];
    const float*         k     = (const float*)d_in[1];
    const float*         v     = (const float*)d_in[2];
    const unsigned char* mask  = (const unsigned char*)d_in[3];  // jnp bool = 1 byte
    const float*         Wq    = (const float*)d_in[4];
    const float*         Wk    = (const float*)d_in[5];
    const float*         Wv    = (const float*)d_in[6];
    const float*         Wfc   = (const float*)d_in[7];
    const float*         bfc   = (const float*)d_in[8];
    const float*         gamma = (const float*)d_in[9];
    const float*         beta  = (const float*)d_in[10];

    const size_t ACT = (size_t)B_ * L_ * D_;        // 2M elements
    const size_t WSZ = (size_t)D_ * D_;             // 256K elements
    unsigned short* p   = (unsigned short*)d_ws;
    unsigned short* Qh   = p;            p += ACT;
    unsigned short* Kh   = p;            p += ACT;
    unsigned short* Vt   = p;            p += ACT;
    unsigned short* ctx  = p;            p += ACT;
    unsigned short* qb   = p;            p += ACT;
    unsigned short* kb   = p;            p += ACT;
    unsigned short* vb   = p;            p += ACT;
    unsigned short* Wqb  = p;            p += WSZ;
    unsigned short* Wkb  = p;            p += WSZ;
    unsigned short* Wvb  = p;            p += WSZ;
    unsigned short* Wfcb = p;            p += WSZ;

    // one-pass f32 -> bf16 conversions (activations + weights)
    const int actN8 = (int)(ACT / 8), wN8 = (int)(WSZ / 8);
    cvt_bf16_kernel<<<(actN8 + 255) / 256, 256, 0, stream>>>(q,   qb,   actN8);
    cvt_bf16_kernel<<<(actN8 + 255) / 256, 256, 0, stream>>>(k,   kb,   actN8);
    cvt_bf16_kernel<<<(actN8 + 255) / 256, 256, 0, stream>>>(v,   vb,   actN8);
    cvt_bf16_kernel<<<(wN8 + 255) / 256,   256, 0, stream>>>(Wq,  Wqb,  wN8);
    cvt_bf16_kernel<<<(wN8 + 255) / 256,   256, 0, stream>>>(Wk,  Wkb,  wN8);
    cvt_bf16_kernel<<<(wN8 + 255) / 256,   256, 0, stream>>>(Wv,  Wvb,  wN8);
    cvt_bf16_kernel<<<(wN8 + 255) / 256,   256, 0, stream>>>(Wfc, Wfcb, wN8);

    proj_kernel<<<512, 128, 0, stream>>>(qb, Wqb, Qh, 0);
    proj_kernel<<<512, 128, 0, stream>>>(kb, Wkb, Kh, 1);
    proj_kernel<<<512, 128, 0, stream>>>(vb, Wvb, Vt, 2);
    attn_kernel<<<512, 128, 0, stream>>>(Qh, Kh, Vt, mask, ctx);
    fc_ln_kernel<<<256, 256, 0, stream>>>(ctx, Wfcb, bfc, q, gamma, beta, (float*)d_out);
}
